// MSFIN_51694226375101
// MI455X (gfx1250) — compile-verified, matrix operand order checked by reading, not run on hardware
//
#include <hip/hip_runtime.h>
#include <hip/hip_bf16.h>
#include <math.h>

#define WAYS 5
#define NQ   75
#define CH   640
#define NT   512
#define CP   64

typedef __attribute__((ext_vector_type(16))) _Float16 v16h;
typedef __attribute__((ext_vector_type(8)))  _Float16 v8h;
typedef __attribute__((ext_vector_type(8)))  float    v8f;

union Frag { v16h v; v8h h[2]; };

// order-preserving float<->uint key for atomicMax on shared memory (ds_max_u32)
__device__ __forceinline__ unsigned fkey(float x) {
  unsigned u = __float_as_uint(x);
  return (u & 0x80000000u) ? ~u : (u | 0x80000000u);
}
__device__ __forceinline__ float funkey(unsigned k) {
  return __uint_as_float((k & 0x80000000u) ? (k & 0x7fffffffu) : ~k);
}

// ---------------------------------------------------------------------------
// Kernel A: center over channels + 1x1 conv (640->64) + BN + ReLU + L2 norm.
// Emits f16 activations token-major [b][512][64] and per-token channel means.
// ---------------------------------------------------------------------------
__global__ __launch_bounds__(256) void proj_kernel(
    const float* __restrict__ x,    // [B][640][512]
    const float* __restrict__ cw,   // [64][640]
    const float* __restrict__ g, const float* __restrict__ be,
    const float* __restrict__ mn, const float* __restrict__ vr,
    _Float16* __restrict__ outF,    // [B][512][64]
    float* __restrict__ muOut)      // [B][512]
{
  const int b  = blockIdx.x >> 4;
  const int t0 = (blockIdx.x & 15) * 32;
  const int tid = threadIdx.x;
  const int tt = tid & 31;   // token within block
  const int cg = tid >> 5;   // 0..7 channel group

  __shared__ float part[8][32];
  __shared__ float mu[32];
  __shared__ float nrm[32];

  const float* xb = x + (size_t)b * CH * NT;

  float s = 0.f;
  for (int c = cg; c < CH; c += 8) s += xb[(size_t)c * NT + t0 + tt];
  part[cg][tt] = s;
  __syncthreads();
  if (cg == 0) {
    float m = 0.f;
#pragma unroll
    for (int j = 0; j < 8; ++j) m += part[j][tt];
    m *= (1.0f / CH);
    mu[tt]  = m;
    nrm[tt] = 0.f;
    muOut[(size_t)b * NT + t0 + tt] = m;
  }
  __syncthreads();

  float acc[8];
#pragma unroll
  for (int j = 0; j < 8; ++j) acc[j] = 0.f;
  const float m = mu[tt];
  for (int c = 0; c < CH; ++c) {
    float xv = xb[(size_t)c * NT + t0 + tt] - m;
#pragma unroll
    for (int j = 0; j < 8; ++j)
      acc[j] = fmaf(xv, cw[(cg * 8 + j) * CH + c], acc[j]);
  }

  float ss = 0.f;
#pragma unroll
  for (int j = 0; j < 8; ++j) {
    int o = cg * 8 + j;
    float sc = g[o] * rsqrtf(vr[o] + 1e-5f);
    float y  = fmaf(acc[j], sc, be[o] - mn[o] * sc);
    y = fmaxf(y, 0.f);
    acc[j] = y;
    ss += y * y;
  }
  atomicAdd(&nrm[tt], ss);
  __syncthreads();

  float inv = 1.0f / fmaxf(sqrtf(nrm[tt]), 1e-8f);
  _Float16* dst = outF + ((size_t)b * NT + t0 + tt) * CP + cg * 8;
#pragma unroll
  for (int j = 0; j < 8; ++j) dst[j] = (_Float16)(acc[j] * inv);
}

// ---------------------------------------------------------------------------
// Kernel B helpers
// ---------------------------------------------------------------------------
__device__ __forceinline__ v8f wmma2(const Frag& a0, const Frag& a1,
                                     const Frag& b0, const Frag& b1) {
  v8f d = {};
  d = __builtin_amdgcn_wmma_f32_16x16x32_f16(false, a0.v, false, b0.v,
                                             (short)0, d, false, false);
  d = __builtin_amdgcn_wmma_f32_16x16x32_f16(false, a1.v, false, b1.v,
                                             (short)0, d, false, false);
  return d;
}

__device__ __forceinline__ void load_qhalf(_Float16* qh, const _Float16* src,
                                           int tid) {
  const v8h* s = (const v8h*)(src + (size_t)tid * CP);
  v8h* d = (v8h*)(qh + tid * CP);
#pragma unroll
  for (int j = 0; j < 8; ++j) d[j] = s[j];
}

__device__ __forceinline__ void load_bfrag(const _Float16* qh, int kt, int n,
                                           int hl, Frag& B0, Frag& B1) {
  const _Float16* bb = qh + (kt * 16 + n) * CP;
  B0.h[0] = *(const v8h*)(bb + 8 * hl);
  B0.h[1] = *(const v8h*)(bb + 16 + 8 * hl);
  B1.h[0] = *(const v8h*)(bb + 32 + 8 * hl);
  B1.h[1] = *(const v8h*)(bb + 48 + 8 * hl);
}

// ---------------------------------------------------------------------------
// Kernel B: per (q,w) pair. corr = spt * qry^T (512x512, K=64) recomputed in
// 3 WMMA passes (stats / partition functions / attention). Row-direction
// statistics are wave-private -> kept entirely in VGPRs, one butterfly per
// pass. Column stats use LDS atomics (cross-wave). Epilogue fuses attention
// pooling + cosine similarity.
// ---------------------------------------------------------------------------
__global__ __launch_bounds__(256) void corr_attn_kernel(
    const _Float16* __restrict__ sptF,  // [5][512][64]
    const _Float16* __restrict__ qryF,  // [75][512][64]
    const float* __restrict__ muS,      // [5][512]
    const float* __restrict__ muQ,      // [75][512]
    const float* __restrict__ tokS,     // [5][640][512]  (raw, un-centered)
    const float* __restrict__ tokQ,     // [75][640][512]
    float* __restrict__ out)            // [75][5]
{
  __shared__ __align__(16) _Float16 qh[256 * CP];  // 32 KB: half of qry tile
  __shared__ float cSum[NT], cSq[NT];              // column stats (k-indexed)
  __shared__ unsigned cMax[NT];
  __shared__ float attnS[NT], attnQ[NT];
  __shared__ float redn[8];

  const int tid  = threadIdx.x;
  const int lane = tid & 31;
  const int wv   = tid >> 5;      // wave 0..7; owns rows [wv*64, wv*64+64)
  const int r    = lane & 15;
  const int hl   = lane >> 4;
  const int n    = r;             // D column within tile
  const int q    = blockIdx.x / WAYS;
  const int w    = blockIdx.x - q * WAYS;

  for (int i = tid; i < NT; i += 256) {
    cSum[i] = 0.f; cSq[i] = 0.f; cMax[i] = 0u; attnQ[i] = 0.f;
  }
  if (tid < 8) redn[tid] = 0.f;

  // Persistent A fragments: wave wv owns i-tiles wv*4 .. wv*4+3.
  Frag A0[4], A1[4];
#pragma unroll
  for (int j = 0; j < 4; ++j) {
    int i = (wv * 4 + j) * 16 + r;
    const _Float16* base = sptF + ((size_t)w * NT + i) * CP;
    A0[j].h[0] = *(const v8h*)(base + 8 * hl);
    A0[j].h[1] = *(const v8h*)(base + 16 + 8 * hl);
    A1[j].h[0] = *(const v8h*)(base + 32 + 8 * hl);
    A1[j].h[1] = *(const v8h*)(base + 48 + 8 * hl);
  }
  __syncthreads();

  const _Float16* qbase = qryF + (size_t)q * NT * CP;

  // ---------------- Pass 1: sum / sumsq / max --------------------------------
  // Row partials live in registers; lane covers columns {256*half + 16*kt + n}.
  float rS[4][8], rS2[4][8], rM[4][8];
#pragma unroll
  for (int j = 0; j < 4; ++j)
#pragma unroll
    for (int v = 0; v < 8; ++v) { rS[j][v] = 0.f; rS2[j][v] = 0.f; rM[j][v] = -3.0e38f; }

  for (int half = 0; half < 2; ++half) {
    load_qhalf(qh, qbase + (size_t)half * 256 * CP, tid);
    __syncthreads();
    for (int kt = 0; kt < 16; ++kt) {
      Frag B0, B1;
      load_bfrag(qh, kt, n, hl, B0, B1);
      const int kg = half * 256 + kt * 16 + n;
      float cs = 0.f, cs2 = 0.f, cm = -3.0e38f;
#pragma unroll
      for (int j = 0; j < 4; ++j) {
        v8f d = wmma2(A0[j], A1[j], B0, B1);
#pragma unroll
        for (int v = 0; v < 8; ++v) {
          float ev = d[v];
          cs += ev; cs2 += ev * ev; cm = fmaxf(cm, ev);
          rS[j][v] += ev; rS2[j][v] += ev * ev; rM[j][v] = fmaxf(rM[j][v], ev);
        }
      }
      cs += __shfl_xor(cs, 16);
      cs2 += __shfl_xor(cs2, 16);
      cm = fmaxf(cm, __shfl_xor(cm, 16));
      if (hl == 0) {
        atomicAdd(&cSum[kg], cs);
        atomicAdd(&cSq[kg], cs2);
        atomicMax(&cMax[kg], fkey(cm));
      }
    }
    __syncthreads();
  }

  // Row finalize: one butterfly per pass; result replicated in every lane of
  // the 16-lane group, which is exactly the lane that needs it later.
  float rScale[4][8], rBias[4][8];  // exp arg = fma(x, rScale, rBias)
#pragma unroll
  for (int j = 0; j < 4; ++j)
#pragma unroll
    for (int v = 0; v < 8; ++v) {
      float s = rS[j][v], s2 = rS2[j][v], mv = rM[j][v];
#pragma unroll
      for (int off = 1; off < 16; off <<= 1) {
        s += __shfl_xor(s, off);
        s2 += __shfl_xor(s2, off);
        mv = fmaxf(mv, __shfl_xor(mv, off));
      }
      float mean = s * (1.0f / NT);
      float var = (s2 - (float)NT * mean * mean) * (1.0f / (NT - 1));
      float sc = 1.0f / (sqrtf(var + 1e-5f) * 5.0f);  // 1/(std * TEMP_ATTN)
      rScale[j][v] = sc;
      rBias[j][v] = -mv * sc;
    }

  // Column finalize (LDS); reuse cSum as sumexp accumulator, cSq as scale.
  for (int i = tid; i < NT; i += 256) {
    float mean = cSum[i] * (1.0f / NT);
    float var = (cSq[i] - (float)NT * mean * mean) * (1.0f / (NT - 1));
    cSq[i] = 1.0f / (sqrtf(var + 1e-5f) * 5.0f);
    cSum[i] = 0.f;
  }
  __syncthreads();

  // ---------------- Pass 2: softmax partition functions ----------------------
  float rZ[4][8];
#pragma unroll
  for (int j = 0; j < 4; ++j)
#pragma unroll
    for (int v = 0; v < 8; ++v) rZ[j][v] = 0.f;

  for (int half = 0; half < 2; ++half) {
    __syncthreads();
    load_qhalf(qh, qbase + (size_t)half * 256 * CP, tid);
    __syncthreads();
    for (int kt = 0; kt < 16; ++kt) {
      Frag B0, B1;
      load_bfrag(qh, kt, n, hl, B0, B1);
      const int kg = half * 256 + kt * 16 + n;
      const float csc = cSq[kg];
      const float cbi = -funkey(cMax[kg]) * csc;
      float se = 0.f;
#pragma unroll
      for (int j = 0; j < 4; ++j) {
        v8f d = wmma2(A0[j], A1[j], B0, B1);
#pragma unroll
        for (int v = 0; v < 8; ++v) {
          float ev = d[v];
          se += __expf(fmaf(ev, csc, cbi));
          rZ[j][v] += __expf(fmaf(ev, rScale[j][v], rBias[j][v]));
        }
      }
      se += __shfl_xor(se, 16);
      if (hl == 0) atomicAdd(&cSum[kg], se);
    }
  }
  // row Z: butterfly once, then invert in registers
#pragma unroll
  for (int j = 0; j < 4; ++j)
#pragma unroll
    for (int v = 0; v < 8; ++v) {
      float z = rZ[j][v];
#pragma unroll
      for (int off = 1; off < 16; off <<= 1) z += __shfl_xor(z, off);
      rZ[j][v] = 1.0f / z;
    }
  __syncthreads();
  for (int i = tid; i < NT; i += 256) cSum[i] = 1.0f / cSum[i];
  __syncthreads();

  // ---------------- Pass 3: attn_s[i] = sum_k p_col; attn_q[k] = sum_i p_row -
  float aS[4][8];
#pragma unroll
  for (int j = 0; j < 4; ++j)
#pragma unroll
    for (int v = 0; v < 8; ++v) aS[j][v] = 0.f;

  for (int half = 0; half < 2; ++half) {
    __syncthreads();
    load_qhalf(qh, qbase + (size_t)half * 256 * CP, tid);
    __syncthreads();
    for (int kt = 0; kt < 16; ++kt) {
      Frag B0, B1;
      load_bfrag(qh, kt, n, hl, B0, B1);
      const int kg = half * 256 + kt * 16 + n;
      const float csc = cSq[kg];
      const float cbi = -funkey(cMax[kg]) * csc;
      const float czr = cSum[kg];
      float sq = 0.f;
#pragma unroll
      for (int j = 0; j < 4; ++j) {
        v8f d = wmma2(A0[j], A1[j], B0, B1);
#pragma unroll
        for (int v = 0; v < 8; ++v) {
          float ev = d[v];
          aS[j][v] += __expf(fmaf(ev, csc, cbi)) * czr;                  // p_col
          sq += __expf(fmaf(ev, rScale[j][v], rBias[j][v])) * rZ[j][v]; // p_row
        }
      }
      sq += __shfl_xor(sq, 16);
      if (hl == 0) atomicAdd(&attnQ[kg], sq);
    }
  }
  // attn_s: butterfly once per row, plain store (rows are wave-exclusive)
#pragma unroll
  for (int j = 0; j < 4; ++j)
#pragma unroll
    for (int v = 0; v < 8; ++v) {
      float a = aS[j][v];
#pragma unroll
      for (int off = 1; off < 16; off <<= 1) a += __shfl_xor(a, off);
      if (n == 0) attnS[(wv * 4 + j) * 16 + 8 * hl + v] = a;
    }
  __syncthreads();

  // ---------------- Epilogue: pooled means + cosine similarity ---------------
  {
    float ps = 0.f, pq = 0.f;
    for (int i = tid; i < NT; i += 256) {
      ps += attnS[i] * muS[(size_t)w * NT + i];
      pq += attnQ[i] * muQ[(size_t)q * NT + i];
    }
    atomicAdd(&redn[3], ps);
    atomicAdd(&redn[4], pq);
  }
  __syncthreads();
  const float amuS = redn[3];
  const float amuQ = redn[4];

  float sdot = 0.f, ssq = 0.f, qsq = 0.f;
  for (int c = tid; c < CH; c += 256) {
    const float* xs = tokS + ((size_t)w * CH + c) * NT;
    const float* xq = tokQ + ((size_t)q * CH + c) * NT;
    float as = 0.f, aq = 0.f;
    for (int i = 0; i < NT; ++i) {
      as = fmaf(attnS[i], xs[i], as);
      aq = fmaf(attnQ[i], xq[i], aq);
    }
    float ps = (as - amuS) * (1.0f / NT);
    float pq = (aq - amuQ) * (1.0f / NT);
    sdot += ps * pq; ssq += ps * ps; qsq += pq * pq;
  }
  atomicAdd(&redn[0], sdot);
  atomicAdd(&redn[1], ssq);
  atomicAdd(&redn[2], qsq);
  __syncthreads();
  if (tid == 0) {
    float ns = fmaxf(sqrtf(redn[1]), 1e-8f);
    float nq = fmaxf(sqrtf(redn[2]), 1e-8f);
    out[q * WAYS + w] = (redn[0] / (ns * nq)) * 5.0f;  // /TEMP = /0.2
  }
}

// ---------------------------------------------------------------------------
extern "C" void kernel_launch(void* const* d_in, const int* in_sizes, int n_in,
                              void* d_out, int out_size, void* d_ws, size_t ws_size,
                              hipStream_t stream) {
  (void)in_sizes; (void)n_in; (void)out_size; (void)ws_size;
  const float* tokS = (const float*)d_in[0];
  const float* tokQ = (const float*)d_in[1];
  const float* cw   = (const float*)d_in[2];
  const float* g    = (const float*)d_in[3];
  const float* be   = (const float*)d_in[4];
  const float* mn   = (const float*)d_in[5];
  const float* vr   = (const float*)d_in[6];
  float* out = (float*)d_out;

  // workspace layout (~5.4 MB)
  _Float16* sptF = (_Float16*)d_ws;                       // 5*512*64 f16
  _Float16* qryF = sptF + (size_t)WAYS * NT * CP;         // 75*512*64 f16
  float* muS = (float*)(qryF + (size_t)NQ * NT * CP);     // 5*512 f32
  float* muQ = muS + WAYS * NT;                           // 75*512 f32

  proj_kernel<<<dim3(WAYS * 16), 256, 0, stream>>>(tokS, cw, g, be, mn, vr,
                                                   sptF, muS);
  proj_kernel<<<dim3(NQ * 16), 256, 0, stream>>>(tokQ, cw, g, be, mn, vr,
                                                 qryF, muQ);
  corr_attn_kernel<<<dim3(NQ * WAYS), 256, 0, stream>>>(sptF, qryF, muS, muQ,
                                                        tokS, tokQ, out);
}
